// Gemma3ForMultimodalLM_60748017435157
// MI455X (gfx1250) — matrix operand order repaired and checked by reading, not compiled, not run on hardware
//
#include <hip/hip_runtime.h>
#include <hip/hip_bf16.h>
#include <math.h>

// ---------------- problem constants ----------------
#define B_    4
#define T_    128
#define S_    8192
#define HID_  2560
#define NH_   8
#define NKV_  4
#define HD_   256
#define INTER_ 10240
#define QKV_N ((NH_ + 2 * NKV_) * HD_)   // 4096
#define M_    (B_ * T_)                   // 512 rows of activations
#define SOFTCAP_ 50.0f
#define SCALING_ 0.0625f                  // 256^-0.5
#define EPS_  1e-6f
#define NEG_  (-2.3819763e+38f)

typedef __bf16 v16bf __attribute__((ext_vector_type(16)));
typedef __bf16 v8bf  __attribute__((ext_vector_type(8)));
typedef __bf16 v4bf  __attribute__((ext_vector_type(4)));
typedef float  v8f   __attribute__((ext_vector_type(8)));

// ============================================================
// WMMA bf16 GEMM:  C[M,N] = A[M,K] * W[N,K]^T   (fp32 in/out)
// Block tile 64(M) x 256(N), K-step 32, 256 threads = 8 waves.
// Wave (wm,wn) in 2x4 grid computes a 32x64 patch = 2x4 WMMA tiles.
// Double-buffered LDS; per iteration: issue 10 global b128 loads for
// tile k+1, run 8 WMMAs on tile k, then convert+store, one barrier.
// LDS layouts make every fragment two contiguous ds_load_b128:
//   A frag lane(m,hl): K runs [8hl,8hl+8) and [16+8hl,24+8hl) of row m
//   B frag lane(n,hl): K runs [8hl,8hl+8) and [16+8hl,24+8hl) of W-row n (no transpose!)
// ============================================================
#define BM 64
#define BN 256
#define BK 32
#define BKP 40   // padded row stride in bf16 (80B: 16B-aligned, bank-spread)

__global__ __launch_bounds__(256) void gemm_bf16_wmma(
    const float* __restrict__ A,   // [M,K] row-major
    const float* __restrict__ W,   // [N,K] row-major (we apply W^T)
    float* __restrict__ C,         // [M,N]
    int M, int N, int K)
{
    __shared__ __attribute__((aligned(16))) __bf16 As[2][BM][BKP];
    __shared__ __attribute__((aligned(16))) __bf16 Bs[2][BN][BKP];

    const int t    = threadIdx.x;
    const int wave = t >> 5;
    const int lane = t & 31;
    const int wm   = wave & 1;          // 0..1  -> M
    const int wn   = wave >> 1;         // 0..3  -> N
    const int m0   = blockIdx.y * BM;
    const int n0   = blockIdx.x * BN;
    const int hl   = lane >> 4;         // half-wave select
    const int l15  = lane & 15;

    v8f acc[2][4];
#pragma unroll
    for (int mi = 0; mi < 2; ++mi)
#pragma unroll
        for (int ni = 0; ni < 4; ++ni)
#pragma unroll
            for (int r = 0; r < 8; ++r) acc[mi][ni][r] = 0.0f;

    // loader coordinates: 8 threads per row, float4 each -> 32 k per row
    const int lr = t >> 3;              // 0..31 row within pass
    const int lc = (t & 7) * 4;         // k offset (float4)
    const float* aPtr = &A[(size_t)(m0 + lr) * K + lc];
    const float* wPtr = &W[(size_t)(n0 + lr) * K + lc];

    float4 areg[2], wreg[8];

    auto loadTile = [&](int k0) {
#pragma unroll
        for (int p = 0; p < 2; ++p)
            areg[p] = *(const float4*)(aPtr + (size_t)p * 32 * K + k0);
#pragma unroll
        for (int p = 0; p < 8; ++p)
            wreg[p] = *(const float4*)(wPtr + (size_t)p * 32 * K + k0);
    };
    auto storeTile = [&](int buf) {
#pragma unroll
        for (int p = 0; p < 2; ++p) {
            v4bf pk = { (__bf16)areg[p].x, (__bf16)areg[p].y,
                        (__bf16)areg[p].z, (__bf16)areg[p].w };
            *(v4bf*)&As[buf][lr + p * 32][lc] = pk;
        }
#pragma unroll
        for (int p = 0; p < 8; ++p) {
            v4bf pk = { (__bf16)wreg[p].x, (__bf16)wreg[p].y,
                        (__bf16)wreg[p].z, (__bf16)wreg[p].w };
            *(v4bf*)&Bs[buf][lr + p * 32][lc] = pk;
        }
    };

    // prologue: stage tile 0
    loadTile(0);
    storeTile(0);
    __syncthreads();

    int cur = 0;
    for (int k0 = 0; k0 < K; k0 += BK) {
        const bool has_next = (k0 + BK) < K;
        if (has_next) loadTile(k0 + BK);   // issue early; waits land later

        // ---- fragments: each = two contiguous 16B LDS loads ----
        v16bf afr[2], bfr[4];
#pragma unroll
        for (int mi = 0; mi < 2; ++mi) {
            const int arow = wm * 32 + mi * 16 + l15;
            const v8bf lo = *(const v8bf*)&As[cur][arow][8 * hl];
            const v8bf hi = *(const v8bf*)&As[cur][arow][16 + 8 * hl];
            afr[mi] = __builtin_shufflevector(lo, hi,
                0,1,2,3,4,5,6,7,8,9,10,11,12,13,14,15);
        }
#pragma unroll
        for (int ni = 0; ni < 4; ++ni) {
            const int brow = wn * 64 + ni * 16 + l15;
            const v8bf lo = *(const v8bf*)&Bs[cur][brow][8 * hl];
            const v8bf hi = *(const v8bf*)&Bs[cur][brow][16 + 8 * hl];
            bfr[ni] = __builtin_shufflevector(lo, hi,
                0,1,2,3,4,5,6,7,8,9,10,11,12,13,14,15);
        }
#pragma unroll
        for (int ni = 0; ni < 4; ++ni)
#pragma unroll
            for (int mi = 0; mi < 2; ++mi)
                acc[mi][ni] = __builtin_amdgcn_wmma_f32_16x16x32_bf16(
                    false, afr[mi], false, bfr[ni],
                    (short)0, acc[mi][ni], false, false);

        if (has_next) storeTile(cur ^ 1);  // write other buffer
        __syncthreads();
        cur ^= 1;
    }

    // ---- store D (VGPR r: M = r + 8*hl, N = lane&15) ----
#pragma unroll
    for (int mi = 0; mi < 2; ++mi)
#pragma unroll
        for (int ni = 0; ni < 4; ++ni) {
            const int col = n0 + wn * 64 + ni * 16 + l15;
#pragma unroll
            for (int r = 0; r < 8; ++r) {
                const int row = m0 + wm * 32 + mi * 16 + hl * 8 + r;
                C[(size_t)row * N + col] = acc[mi][ni][r];
            }
        }
}

// ============================================================
// Row RMSNorm:  y = x * rsqrt(mean(x^2)+eps) * (1+w)     [row len L]
// optional residual add (res != nullptr): y = res + rmsnorm(x)
// ============================================================
__global__ __launch_bounds__(256) void rmsnorm_kernel(
    const float* __restrict__ x, const float* __restrict__ w,
    const float* __restrict__ res, float* __restrict__ y, int L)
{
    const size_t base = (size_t)blockIdx.x * L;
    const int t = threadIdx.x, lane = t & 31, wave = t >> 5;
    float ss = 0.f;
    for (int i = t; i < L; i += 256) { float v = x[base + i]; ss += v * v; }
#pragma unroll
    for (int off = 16; off > 0; off >>= 1) ss += __shfl_xor(ss, off, 32);
    __shared__ float wsum[8];
    if (lane == 0) wsum[wave] = ss;
    __syncthreads();
    float tot = 0.f;
#pragma unroll
    for (int i = 0; i < 8; ++i) tot += wsum[i];
    const float rs = rsqrtf(tot / (float)L + EPS_);
    for (int i = t; i < L; i += 256) {
        float o = x[base + i] * rs * (1.f + w[i]);
        if (res) o += res[base + i];
        y[base + i] = o;
    }
}

// ============================================================
// Q/K post-processing: per-head RMSNorm (HD=256) + RoPE (+SCALING for Q)
// grid: (B*T, NH+NKV), block: 256 threads (one per channel)
// ============================================================
__global__ __launch_bounds__(256) void qk_post_kernel(
    const float* __restrict__ qkv, const float* __restrict__ cosb,
    const float* __restrict__ sinb, const float* __restrict__ qw,
    const float* __restrict__ kw, float* __restrict__ qbuf,
    float* __restrict__ kbuf)
{
    const int bt = blockIdx.x;                 // b*T + t
    const int hh = blockIdx.y;                 // 0..NH+NKV-1
    const int tt = bt & (T_ - 1);
    const int d  = threadIdx.x;
    const bool isq = hh < NH_;
    const float* src = qkv + (size_t)bt * QKV_N +
                       (isq ? hh * HD_ : NH_ * HD_ + (hh - NH_) * HD_);
    const float x = src[d];

    __shared__ float red[256];
    __shared__ float row[256];
    red[d] = x * x;
    __syncthreads();
#pragma unroll
    for (int s = 128; s > 0; s >>= 1) {
        if (d < s) red[d] += red[d + s];
        __syncthreads();
    }
    const float rs = rsqrtf(red[0] / (float)HD_ + EPS_);
    const float* wv = isq ? qw : kw;
    const float xn = x * rs * (1.f + wv[d]);
    row[d] = xn;
    __syncthreads();

    float o;
    if (d < 128) {
        const float c = cosb[tt * 128 + d], s = sinb[tt * 128 + d];
        o = xn * c - row[d + 128] * s;
    } else {
        const int dd = d - 128;
        const float c = cosb[tt * 128 + dd], s = sinb[tt * 128 + dd];
        o = row[dd] * s + xn * c;
    }
    if (isq) qbuf[((size_t)bt * NH_ + hh) * HD_ + d] = o * SCALING_;
    else     kbuf[((size_t)bt * NKV_ + (hh - NH_)) * HD_ + d] = o;
}

// ============================================================
// Attention: only keys 0..127 are live (cache written at 0..T-1, causal
// mask kills the rest; WINDOW=1024 > 127 so no truncation).
// One block = (16 query rows, head h, batch b). Softcap tanh + causal
// softmax in fp32, then P*V. 256 threads = 8 waves; wave handles 2 rows.
// ============================================================
__global__ __launch_bounds__(256) void attn_kernel(
    const float* __restrict__ qbuf, const float* __restrict__ kbuf,
    const float* __restrict__ qkv, float* __restrict__ attn_out)
{
    const int q0  = blockIdx.x * 16;
    const int h   = blockIdx.y;
    const int b   = blockIdx.z;
    const int kvh = h >> 1;                     // g = NH/NKV = 2
    const int t    = threadIdx.x;
    const int wave = t >> 5, lane = t & 31;

    __shared__ float sQ[16][HD_ + 4];
    __shared__ float sS[16][T_];

    for (int i = t; i < 16 * HD_; i += 256) {
        const int qi = i >> 8, d = i & (HD_ - 1);
        sQ[qi][d] = qbuf[((size_t)(b * T_ + q0 + qi) * NH_ + h) * HD_ + d];
    }
    __syncthreads();

    // scores
    for (int j = 0; j < T_; ++j) {
        const float* kp = kbuf + ((size_t)(b * T_ + j) * NKV_ + kvh) * HD_;
        float kr[8];
#pragma unroll
        for (int i = 0; i < 8; ++i) kr[i] = kp[lane + 32 * i];
#pragma unroll
        for (int qr = 0; qr < 2; ++qr) {
            const int qi = wave * 2 + qr;
            float s = 0.f;
#pragma unroll
            for (int i = 0; i < 8; ++i) s += kr[i] * sQ[qi][lane + 32 * i];
#pragma unroll
            for (int off = 16; off > 0; off >>= 1) s += __shfl_xor(s, off, 32);
            if (lane == 0) sS[qi][j] = s;
        }
    }
    __syncthreads();

    // softcap + causal mask + softmax (row-wise, fp32)
#pragma unroll
    for (int qr = 0; qr < 2; ++qr) {
        const int qi = wave * 2 + qr;
        const int qpos = q0 + qi;
        float vals[4], mx = NEG_;
#pragma unroll
        for (int i = 0; i < 4; ++i) {
            const int j = lane + 32 * i;
            float s = SOFTCAP_ * tanhf(sS[qi][j] * (1.f / SOFTCAP_));
            if (j > qpos) s = NEG_;
            vals[i] = s;
            mx = fmaxf(mx, s);
        }
#pragma unroll
        for (int off = 16; off > 0; off >>= 1) mx = fmaxf(mx, __shfl_xor(mx, off, 32));
        float sum = 0.f;
#pragma unroll
        for (int i = 0; i < 4; ++i) { vals[i] = __expf(vals[i] - mx); sum += vals[i]; }
#pragma unroll
        for (int off = 16; off > 0; off >>= 1) sum += __shfl_xor(sum, off, 32);
        const float inv = 1.f / sum;
#pragma unroll
        for (int i = 0; i < 4; ++i) sS[qi][lane + 32 * i] = vals[i] * inv;
    }
    __syncthreads();

    // out[qi][d] = sum_j p[qi][j] * V[j][d]   (thread <-> d)
    float acc[16];
#pragma unroll
    for (int qi = 0; qi < 16; ++qi) acc[qi] = 0.f;
    const float* vbase = qkv + (size_t)(b * T_) * QKV_N +
                         (NH_ + NKV_) * HD_ + kvh * HD_ + t;
    for (int j = 0; j < T_; ++j) {
        const float vj = vbase[(size_t)j * QKV_N];
#pragma unroll
        for (int qi = 0; qi < 16; ++qi) acc[qi] += sS[qi][j] * vj;
    }
#pragma unroll
    for (int qi = 0; qi < 16; ++qi)
        attn_out[((size_t)(b * T_ + q0 + qi) * NH_ + h) * HD_ + t] = acc[qi];
}

// ============================================================
// GEGLU elementwise: inter = gelu_tanh(gate) * up
// ============================================================
__global__ __launch_bounds__(256) void gelu_mul_kernel(
    const float* __restrict__ gate, const float* __restrict__ up,
    float* __restrict__ out, size_t n)
{
    for (size_t i = (size_t)blockIdx.x * 256 + threadIdx.x; i < n;
         i += (size_t)gridDim.x * 256) {
        const float g = gate[i];
        const float gg = 0.5f * g *
            (1.f + tanhf(0.7978845608028654f * (g + 0.044715f * g * g * g)));
        out[i] = gg * up[i];
    }
}

// ============================================================
// Host launcher
// ============================================================
extern "C" void kernel_launch(void* const* d_in, const int* in_sizes, int n_in,
                              void* d_out, int out_size, void* d_ws, size_t ws_size,
                              hipStream_t stream) {
    const float* hidden   = (const float*)d_in[0];
    const float* cosb     = (const float*)d_in[1];
    const float* sinb     = (const float*)d_in[2];
    const float* w_qkv    = (const float*)d_in[8];
    const float* w_o      = (const float*)d_in[9];
    const float* q_norm_w = (const float*)d_in[10];
    const float* k_norm_w = (const float*)d_in[11];
    const float* in_ln    = (const float*)d_in[12];
    const float* post_attn= (const float*)d_in[13];
    const float* pre_ffw  = (const float*)d_in[14];
    const float* post_ffw = (const float*)d_in[15];
    const float* w_gate   = (const float*)d_in[16];
    const float* w_up     = (const float*)d_in[17];
    const float* w_down   = (const float*)d_in[18];
    float* out = (float*)d_out;

    float* ws = (float*)d_ws;
    float* h_norm = ws;                              // 512*2560
    float* qkv    = h_norm + (size_t)M_ * HID_;      // 512*4096
    float* qbuf   = qkv    + (size_t)M_ * QKV_N;     // 512*2048
    float* kbuf   = qbuf   + (size_t)M_ * NH_ * HD_; // 512*1024
    float* attn   = kbuf   + (size_t)M_ * NKV_ * HD_;// 512*2048
    float* oproj  = attn   + (size_t)M_ * NH_ * HD_; // 512*2560
    float* h2     = oproj  + (size_t)M_ * HID_;      // 512*2560
    float* x2     = h2     + (size_t)M_ * HID_;      // 512*2560
    float* gate   = x2     + (size_t)M_ * HID_;      // 512*10240
    float* up     = gate   + (size_t)M_ * INTER_;    // 512*10240
    float* mlp    = up     + (size_t)M_ * INTER_;    // 512*2560

    // 1. input RMSNorm
    rmsnorm_kernel<<<M_, 256, 0, stream>>>(hidden, in_ln, nullptr, h_norm, HID_);
    // 2. QKV projection (WMMA)
    gemm_bf16_wmma<<<dim3(QKV_N / BN, M_ / BM), 256, 0, stream>>>(
        h_norm, w_qkv, qkv, M_, QKV_N, HID_);
    // 3. per-head norm + RoPE (+scale for Q)
    qk_post_kernel<<<dim3(M_, NH_ + NKV_), 256, 0, stream>>>(
        qkv, cosb, sinb, q_norm_w, k_norm_w, qbuf, kbuf);
    // 4. attention
    attn_kernel<<<dim3(T_ / 16, NH_, B_), 256, 0, stream>>>(qbuf, kbuf, qkv, attn);
    // 5. output projection (WMMA)
    gemm_bf16_wmma<<<dim3(HID_ / BN, M_ / BM), 256, 0, stream>>>(
        attn, w_o, oproj, M_, HID_, NH_ * HD_);
    // 6. h = residual + rmsnorm(oproj)
    rmsnorm_kernel<<<M_, 256, 0, stream>>>(oproj, post_attn, hidden, h2, HID_);
    // 7. x = rmsnorm(h)
    rmsnorm_kernel<<<M_, 256, 0, stream>>>(h2, pre_ffw, nullptr, x2, HID_);
    // 8-9. gate & up projections (WMMA)
    gemm_bf16_wmma<<<dim3(INTER_ / BN, M_ / BM), 256, 0, stream>>>(
        x2, w_gate, gate, M_, INTER_, HID_);
    gemm_bf16_wmma<<<dim3(INTER_ / BN, M_ / BM), 256, 0, stream>>>(
        x2, w_up, up, M_, INTER_, HID_);
    // 10. GEGLU elementwise (write into gate buffer)
    gelu_mul_kernel<<<4096, 256, 0, stream>>>(gate, up, gate, (size_t)M_ * INTER_);
    // 11. down projection (WMMA)
    gemm_bf16_wmma<<<dim3(HID_ / BN, M_ / BM), 256, 0, stream>>>(
        gate, w_down, mlp, M_, HID_, INTER_);
    // 12. out = h + rmsnorm(mlp)
    rmsnorm_kernel<<<M_, 256, 0, stream>>>(mlp, post_ffw, h2, out, HID_);
}
